// TransformerBlock_48421461295130
// MI455X (gfx1250) — compile-verified, matrix operand order checked by reading, not run on hardware
//
#include <hip/hip_runtime.h>
#include <hip/hip_bf16.h>
#include <math.h>

// ---------------------------------------------------------------------------
// LLaMA transformer block for MI455X (gfx1250, wave32, WMMA).
// Compute-bound (~0.9 TFLOP vs ~50us of HBM traffic at 23.3 TB/s), so all
// GEMMs use v_wmma_f32_16x16x32_bf16 with fp32 accumulators; reductions
// (rmsnorm/softmax/rope) stay fp32 on VALU. GEMM uses register-staged
// double buffering so global latency overlaps the WMMA pipe.
// ---------------------------------------------------------------------------

#define SEQ    2048
#define DIM    4096
#define NHEADS 32
#define HDIM   128
#define HIDDEN 11008

typedef __bf16 bf16_t;
typedef __attribute__((ext_vector_type(16))) __bf16 v16bf;
typedef __attribute__((ext_vector_type(8)))  __bf16 v8bf;
typedef __attribute__((ext_vector_type(8)))  float  v8f;

// ---------------------------------------------------------------------------
// fp32 -> bf16 cast (weights), grid-stride
// ---------------------------------------------------------------------------
__global__ __launch_bounds__(256) void cast_bf16_kernel(const float* __restrict__ src,
                                                        bf16_t* __restrict__ dst,
                                                        long n) {
  long i = (long)blockIdx.x * blockDim.x + threadIdx.x;
  long stride = (long)gridDim.x * blockDim.x;
  for (; i < n; i += stride) dst[i] = (bf16_t)src[i];
}

// ---------------------------------------------------------------------------
// RMSNorm over DIM, output bf16. One block per row.
// ---------------------------------------------------------------------------
__global__ __launch_bounds__(256) void rmsnorm_kernel(const float* __restrict__ x,
                                                      const float* __restrict__ w,
                                                      bf16_t* __restrict__ out) {
  __shared__ float red[256];
  int tid = threadIdx.x;
  const float* xr = x + (size_t)blockIdx.x * DIM;
  float ss = 0.f;
  for (int i = tid; i < DIM; i += 256) { float v = xr[i]; ss += v * v; }
  red[tid] = ss;
  __syncthreads();
  for (int s = 128; s > 0; s >>= 1) {
    if (tid < s) red[tid] += red[tid + s];
    __syncthreads();
  }
  float inv = rsqrtf(red[0] / (float)DIM + 1e-5f);
  bf16_t* orow = out + (size_t)blockIdx.x * DIM;
  for (int i = tid; i < DIM; i += 256) orow[i] = (bf16_t)(xr[i] * inv * w[i]);
}

// ---------------------------------------------------------------------------
// RoPE: fp32 in -> bf16 out. One thread per complex pair.
// ---------------------------------------------------------------------------
__global__ __launch_bounds__(256) void rope_kernel(const float* __restrict__ in,
                                                   const float* __restrict__ cs,
                                                   const float* __restrict__ sn,
                                                   bf16_t* __restrict__ out) {
  int idx = blockIdx.x * blockDim.x + threadIdx.x;   // SEQ*NHEADS*64 threads
  if (idx >= SEQ * NHEADS * (HDIM / 2)) return;
  int p = idx & 63;
  int head = (idx >> 6) & 31;
  int s = idx >> 11;
  float c = cs[s * 64 + p];
  float si = sn[s * 64 + p];
  size_t base = (size_t)s * DIM + head * HDIM + 2 * p;
  float xr = in[base], xi = in[base + 1];
  out[base]     = (bf16_t)(xr * c - xi * si);
  out[base + 1] = (bf16_t)(xr * si + xi * c);
}

// ---------------------------------------------------------------------------
// Row softmax (N cols), fp32 scores (mask pre-applied) -> bf16 probs.
// ---------------------------------------------------------------------------
__global__ __launch_bounds__(256) void softmax_kernel(const float* __restrict__ S,
                                                      bf16_t* __restrict__ P, int N) {
  __shared__ float red[256];
  int tid = threadIdx.x;
  const float* sr = S + (size_t)blockIdx.x * N;
  float mx = -3.4e38f;
  for (int i = tid; i < N; i += 256) mx = fmaxf(mx, sr[i]);
  red[tid] = mx;
  __syncthreads();
  for (int s = 128; s > 0; s >>= 1) {
    if (tid < s) red[tid] = fmaxf(red[tid], red[tid + s]);
    __syncthreads();
  }
  mx = red[0];
  __syncthreads();
  float sum = 0.f;
  for (int i = tid; i < N; i += 256) sum += __expf(sr[i] - mx);
  red[tid] = sum;
  __syncthreads();
  for (int s = 128; s > 0; s >>= 1) {
    if (tid < s) red[tid] += red[tid + s];
    __syncthreads();
  }
  float inv = 1.f / red[0];
  bf16_t* pr = P + (size_t)blockIdx.x * N;
  for (int i = tid; i < N; i += 256) pr[i] = (bf16_t)(__expf(sr[i] - mx) * inv);
}

// ---------------------------------------------------------------------------
// Tiled bf16 WMMA GEMM:  C[M,N] = A[M,K] * B[K,N]  (+ epilogue)
//   EPI 0: C fp32
//   EPI 1: C bf16
//   EPI 2: C bf16 = silu(aux) * acc          (SwiGLU gate fuse)
//   EPI 3: C fp32 = acc*scale + causal mask  (scores)
//   EPI 4: C fp32 = aux + acc                (residual add)
// TRANSB: B stored row-major N x K (ldb = row stride) -> computes A*B^T.
// Block tile 128x128x32, 256 threads = 8 waves, wave tile 64x32.
// Register-staged + LDS double buffering: next tile's global loads are
// issued before the WMMA phase, so their waitcnt lands after the math.
// ---------------------------------------------------------------------------
template <int EPI, bool TRANSB>
__global__ __launch_bounds__(256) void gemm_wmma(const bf16_t* __restrict__ A,
                                                 const bf16_t* __restrict__ B,
                                                 void* __restrict__ Cout,
                                                 const float* __restrict__ aux,
                                                 int M, int N, int K,
                                                 int lda, int ldb, int ldc,
                                                 float scale) {
  constexpr int BM = 128, BN = 128, BK = 32;
  constexpr int LSTR = BK + 8;            // 40 bf16 = 80 B row stride (16B-aligned)
  __shared__ bf16_t As[2][BM * LSTR];
  __shared__ bf16_t Bs[2][BN * LSTR];

  const int tid  = threadIdx.x;
  const int bm   = blockIdx.y * BM;
  const int bn   = blockIdx.x * BN;
  const int wave = tid >> 5;
  const int lane = tid & 31;
  const int wrow = wave >> 2;             // 0..1 -> M offset 64*wrow
  const int wcol = wave & 3;              // 0..3 -> N offset 32*wcol
  const int lr   = lane & 15;
  const int kb   = (lane >> 4) * 8;       // ISA 16-bit frag layout: K half select

  // ---- per-thread staging addresses (running pointers, no per-iter mul) ----
  const bf16_t* aSrc[2];
  const bf16_t* bSrc[2];
  int aDst[2], bDst[2];
  size_t bStep;
#pragma unroll
  for (int r = 0; r < 2; ++r) {
    int c  = r * 256 + tid;               // 512 chunks of 8 bf16 per tile
    int m  = c >> 2;
    int kk = (c & 3) * 8;
    aSrc[r] = A + (size_t)(bm + m) * lda + kk;
    aDst[r] = m * LSTR + kk;
  }
  if (TRANSB) {
#pragma unroll
    for (int r = 0; r < 2; ++r) {
      int c  = r * 256 + tid;
      int n  = c >> 2;
      int kk = (c & 3) * 8;
      bSrc[r] = B + (size_t)(bn + n) * ldb + kk;
      bDst[r] = n * LSTR + kk;
    }
    bStep = BK;
  } else {
#pragma unroll
    for (int r = 0; r < 2; ++r) {
      int c  = r * 256 + tid;
      int kk = c >> 4;                    // 0..31
      int n  = (c & 15) * 8;
      bSrc[r] = B + (size_t)kk * ldb + bn + n;
      bDst[r] = n * LSTR + kk;            // scatter: +i*LSTR
    }
    bStep = (size_t)BK * ldb;
  }

  v8f  acc[4][2] = {};
  v8bf aReg[2], bReg[2];

  // ---- prologue: fetch + stage tile 0 into buffer 0 ----
#pragma unroll
  for (int r = 0; r < 2; ++r) {
    aReg[r] = *(const v8bf*)(aSrc[r]);
    bReg[r] = *(const v8bf*)(bSrc[r]);
    aSrc[r] += BK;
    bSrc[r] += bStep;
  }
#pragma unroll
  for (int r = 0; r < 2; ++r) {
    *(v8bf*)(&As[0][aDst[r]]) = aReg[r];
    if (TRANSB) {
      *(v8bf*)(&Bs[0][bDst[r]]) = bReg[r];
    } else {
#pragma unroll
      for (int i = 0; i < 8; ++i) Bs[0][bDst[r] + i * LSTR] = bReg[r][i];
    }
  }
  __syncthreads();

  const int nsteps = K / BK;
  for (int s = 0; s < nsteps; ++s) {
    const int cur = s & 1;
    // ---- issue next tile's global loads (wait deferred past the WMMAs) ----
    if (s + 1 < nsteps) {
#pragma unroll
      for (int r = 0; r < 2; ++r) {
        aReg[r] = *(const v8bf*)(aSrc[r]);
        bReg[r] = *(const v8bf*)(bSrc[r]);
        aSrc[r] += BK;
        bSrc[r] += bStep;
      }
      if (s + 2 < nsteps) __builtin_prefetch(aSrc[0], 0, 0);  // global_prefetch_b8
    }

    // ---- fragments from LDS[cur], per ISA 16-bit A/B layout ----
    v16bf afrag[4], bfrag[2];
#pragma unroll
    for (int i = 0; i < 4; ++i) {
      int m = wrow * 64 + i * 16 + lr;
      v8bf lo = *(const v8bf*)(&As[cur][m * LSTR + kb]);
      v8bf hi = *(const v8bf*)(&As[cur][m * LSTR + kb + 16]);
      afrag[i] = __builtin_shufflevector(lo, hi, 0, 1, 2, 3, 4, 5, 6, 7,
                                         8, 9, 10, 11, 12, 13, 14, 15);
    }
#pragma unroll
    for (int j = 0; j < 2; ++j) {
      int n = wcol * 32 + j * 16 + lr;
      v8bf lo = *(const v8bf*)(&Bs[cur][n * LSTR + kb]);
      v8bf hi = *(const v8bf*)(&Bs[cur][n * LSTR + kb + 16]);
      bfrag[j] = __builtin_shufflevector(lo, hi, 0, 1, 2, 3, 4, 5, 6, 7,
                                         8, 9, 10, 11, 12, 13, 14, 15);
    }
#pragma unroll
    for (int i = 0; i < 4; ++i)
#pragma unroll
      for (int j = 0; j < 2; ++j)
        acc[i][j] = __builtin_amdgcn_wmma_f32_16x16x32_bf16(
            false, afrag[i], false, bfrag[j], (short)0, acc[i][j], false, false);

    // ---- stage next tile into the alternate buffer ----
    if (s + 1 < nsteps) {
      const int nxt = cur ^ 1;
#pragma unroll
      for (int r = 0; r < 2; ++r) {
        *(v8bf*)(&As[nxt][aDst[r]]) = aReg[r];
        if (TRANSB) {
          *(v8bf*)(&Bs[nxt][bDst[r]]) = bReg[r];
        } else {
#pragma unroll
          for (int i = 0; i < 8; ++i) Bs[nxt][bDst[r] + i * LSTR] = bReg[r][i];
        }
      }
    }
    __syncthreads();
  }

  // ---- epilogue: C/D layout: VGPR r -> M = r + 8*(lane>=16), N = lane%16 ----
  const int madd = (lane >> 4) * 8;
#pragma unroll
  for (int i = 0; i < 4; ++i) {
#pragma unroll
    for (int j = 0; j < 2; ++j) {
      int ng = bn + wcol * 32 + j * 16 + lr;
#pragma unroll
      for (int r = 0; r < 8; ++r) {
        int mg = bm + wrow * 64 + i * 16 + madd + r;
        float v = acc[i][j][r];
        size_t cidx = (size_t)mg * ldc + ng;
        if (EPI == 0) {
          ((float*)Cout)[cidx] = v;
        } else if (EPI == 1) {
          ((bf16_t*)Cout)[cidx] = (bf16_t)v;
        } else if (EPI == 2) {
          float g = aux[cidx];
          float sg = g / (1.f + __expf(-g));
          ((bf16_t*)Cout)[cidx] = (bf16_t)(sg * v);
        } else if (EPI == 3) {
          ((float*)Cout)[cidx] = v * scale + ((ng > mg) ? -1e9f : 0.f);
        } else if (EPI == 4) {
          ((float*)Cout)[cidx] = aux[cidx] + v;
        }
      }
    }
  }
}

// ---------------------------------------------------------------------------
// Host-side orchestration
// ---------------------------------------------------------------------------
#define GEMM(EPI, TB, A, B, C, AUX, M_, N_, K_, LDA, LDB, LDC, SC)                  \
  gemm_wmma<EPI, TB><<<dim3((N_) / 128, (M_) / 128), 256, 0, stream>>>(             \
      (const bf16_t*)(A), (const bf16_t*)(B), (void*)(C), (const float*)(AUX),      \
      (M_), (N_), (K_), (LDA), (LDB), (LDC), (SC))

extern "C" void kernel_launch(void* const* d_in, const int* in_sizes, int n_in,
                              void* d_out, int out_size, void* d_ws, size_t ws_size,
                              hipStream_t stream) {
  const float* x           = (const float*)d_in[0];
  const float* w_q         = (const float*)d_in[1];
  const float* w_k         = (const float*)d_in[2];
  const float* w_v         = (const float*)d_in[3];
  const float* w_o         = (const float*)d_in[4];
  const float* w1          = (const float*)d_in[5];
  const float* w2          = (const float*)d_in[6];
  const float* w3          = (const float*)d_in[7];
  const float* attn_norm_w = (const float*)d_in[8];
  const float* ffn_norm_w  = (const float*)d_in[9];
  const float* freqs_cos   = (const float*)d_in[10];
  const float* freqs_sin   = (const float*)d_in[11];
  // d_in[12] = mask (causal mask applied analytically), d_in[13] = start_pos (0)
  float* out = (float*)d_out;

  // --- workspace carve-up ---
  char* p = (char*)d_ws;
  auto carve = [&](size_t bytes) -> void* {
    void* r = (void*)p;
    p += (bytes + 255) & ~(size_t)255;
    return r;
  };
  bf16_t* wqb   = (bf16_t*)carve((size_t)DIM * DIM * 2);
  bf16_t* wkb   = (bf16_t*)carve((size_t)DIM * DIM * 2);
  bf16_t* wvb   = (bf16_t*)carve((size_t)DIM * DIM * 2);
  bf16_t* wob   = (bf16_t*)carve((size_t)DIM * DIM * 2);
  bf16_t* w1b   = (bf16_t*)carve((size_t)DIM * HIDDEN * 2);
  bf16_t* w3b   = (bf16_t*)carve((size_t)DIM * HIDDEN * 2);
  bf16_t* w2b   = (bf16_t*)carve((size_t)HIDDEN * DIM * 2);
  bf16_t* hb    = (bf16_t*)carve((size_t)SEQ * DIM * 2);   // rmsnorm(x)
  float*  Qf    = (float*)carve((size_t)SEQ * DIM * 4);
  float*  Kf    = (float*)carve((size_t)SEQ * DIM * 4);
  bf16_t* qb    = (bf16_t*)carve((size_t)SEQ * DIM * 2);   // post-RoPE
  bf16_t* kb    = (bf16_t*)carve((size_t)SEQ * DIM * 2);
  bf16_t* vb    = (bf16_t*)carve((size_t)SEQ * DIM * 2);
  bf16_t* attnb = (bf16_t*)carve((size_t)SEQ * DIM * 2);
  float*  scrS  = (float*)carve((size_t)SEQ * SEQ * 4);    // per-head scores
  bf16_t* scrP  = (bf16_t*)carve((size_t)SEQ * SEQ * 2);   // per-head probs
  float*  hbuf  = (float*)carve((size_t)SEQ * DIM * 4);    // x + attn@w_o
  bf16_t* hffb  = (bf16_t*)carve((size_t)SEQ * DIM * 2);   // rmsnorm(h)
  float*  g1    = (float*)carve((size_t)SEQ * HIDDEN * 4); // h_ff @ w1
  bf16_t* ffb   = (bf16_t*)carve((size_t)SEQ * HIDDEN * 2);
  (void)ws_size; (void)in_sizes; (void)n_in; (void)out_size;

  // --- 1. cast weights to bf16 ---
  cast_bf16_kernel<<<4096, 256, 0, stream>>>(w_q, wqb, (long)DIM * DIM);
  cast_bf16_kernel<<<4096, 256, 0, stream>>>(w_k, wkb, (long)DIM * DIM);
  cast_bf16_kernel<<<4096, 256, 0, stream>>>(w_v, wvb, (long)DIM * DIM);
  cast_bf16_kernel<<<4096, 256, 0, stream>>>(w_o, wob, (long)DIM * DIM);
  cast_bf16_kernel<<<4096, 256, 0, stream>>>(w1, w1b, (long)DIM * HIDDEN);
  cast_bf16_kernel<<<4096, 256, 0, stream>>>(w3, w3b, (long)DIM * HIDDEN);
  cast_bf16_kernel<<<4096, 256, 0, stream>>>(w2, w2b, (long)HIDDEN * DIM);

  // --- 2. attention input norm ---
  rmsnorm_kernel<<<SEQ, 256, 0, stream>>>(x, attn_norm_w, hb);

  // --- 3. Q/K/V projections (bf16 WMMA) ---
  GEMM(0, false, hb, wqb, Qf, nullptr, SEQ, DIM, DIM, DIM, DIM, DIM, 1.f);
  GEMM(0, false, hb, wkb, Kf, nullptr, SEQ, DIM, DIM, DIM, DIM, DIM, 1.f);
  GEMM(1, false, hb, wvb, vb, nullptr, SEQ, DIM, DIM, DIM, DIM, DIM, 1.f);

  // --- 4. RoPE on Q and K (fp32 -> bf16) ---
  int rope_threads = SEQ * NHEADS * (HDIM / 2);
  rope_kernel<<<(rope_threads + 255) / 256, 256, 0, stream>>>(Qf, freqs_cos, freqs_sin, qb);
  rope_kernel<<<(rope_threads + 255) / 256, 256, 0, stream>>>(Kf, freqs_cos, freqs_sin, kb);

  // --- 5. attention, head by head (scores buffer reused; stream-serialized) ---
  const float inv_sqrt_d = 0.088388347648318447f;  // 1/sqrt(128)
  for (int h = 0; h < NHEADS; ++h) {
    // S = Q_h @ K_h^T * scale + causal mask   (TRANSB: contraction over head_dim)
    GEMM(3, true, qb + h * HDIM, kb + h * HDIM, scrS, nullptr,
         SEQ, SEQ, HDIM, DIM, DIM, SEQ, inv_sqrt_d);
    softmax_kernel<<<SEQ, 256, 0, stream>>>(scrS, scrP, SEQ);
    // attn_h = P @ V_h  -> attnb[:, h*128 : h*128+128]
    GEMM(1, false, scrP, vb + h * HDIM, attnb + h * HDIM, nullptr,
         SEQ, HDIM, SEQ, SEQ, DIM, DIM, 1.f);
  }

  // --- 6. h = x + attn @ w_o ---
  GEMM(4, false, attnb, wob, hbuf, x, SEQ, DIM, DIM, DIM, DIM, DIM, 1.f);

  // --- 7. FFN: out = h + (silu(h_ff@w1) * (h_ff@w3)) @ w2 ---
  rmsnorm_kernel<<<SEQ, 256, 0, stream>>>(hbuf, ffn_norm_w, hffb);
  GEMM(0, false, hffb, w1b, g1, nullptr, SEQ, HIDDEN, DIM, DIM, HIDDEN, HIDDEN, 1.f);
  GEMM(2, false, hffb, w3b, ffb, g1, SEQ, HIDDEN, DIM, DIM, HIDDEN, HIDDEN, 1.f);
  GEMM(4, false, ffb, w2b, out, hbuf, SEQ, DIM, HIDDEN, HIDDEN, DIM, DIM, 1.f);
}
